// ClientModel_9560597200898
// MI455X (gfx1250) — matrix operand order repaired and checked by reading, not compiled, not run on hardware
//
#include <hip/hip_runtime.h>

// ---------------------------------------------------------------------------
// Problem constants (from reference): B=4096, T=80, V=80, E=8, H=256
// ---------------------------------------------------------------------------
#define BATCH   4096
#define SEQT    80
#define VOCAB   80
#define EMBD    8
#define HID     256
#define FOURH   1024

typedef __attribute__((ext_vector_type(16))) _Float16 v16h;
typedef __attribute__((ext_vector_type(8)))  _Float16 v8h;
typedef __attribute__((ext_vector_type(8)))  float    v8f;

static __device__ __forceinline__ v8f zero8() {
    v8f z;
#pragma unroll
    for (int i = 0; i < 8; ++i) z[i] = 0.0f;
    return z;
}

static __device__ __forceinline__ float sigf(float x) {
    return 1.0f / (1.0f + __expf(-x));
}

// ---------------------------------------------------------------------------
// Prep: transpose + f16-convert recurrent/input weight blocks.
//   W1hT[n][k] = W1[8+k][n]     (layer-1 hidden weights,  [1024][256] f16)
//   W2xT[n][k] = W2[k][n]       (layer-2 input  weights,  [1024][256] f16)
//   W2hT[n][k] = W2[256+k][n]   (layer-2 hidden weights,  [1024][256] f16)
// Transposed layout makes each WMMA B-fragment two contiguous 16B loads.
// ---------------------------------------------------------------------------
__global__ void prep_weights_kernel(const float* __restrict__ W1,
                                    const float* __restrict__ W2,
                                    _Float16* __restrict__ W1hT,
                                    _Float16* __restrict__ W2xT,
                                    _Float16* __restrict__ W2hT) {
    int idx = blockIdx.x * blockDim.x + threadIdx.x;   // 0 .. 1024*256-1
    if (idx >= FOURH * HID) return;
    int n = idx >> 8;     // gate column 0..1023
    int k = idx & 255;    // K row 0..255
    W1hT[n * HID + k] = (_Float16)W1[(EMBD + k) * FOURH + n];
    W2xT[n * HID + k] = (_Float16)W2[k * FOURH + n];
    W2hT[n * HID + k] = (_Float16)W2[(HID + k) * FOURH + n];
}

// table1[v][n] = b1[n] + sum_e emb[v][e] * W1[e][n]   (V=80 rows -> x-GEMM becomes lookup)
__global__ void build_table_kernel(const float* __restrict__ emb,
                                   const float* __restrict__ W1,
                                   const float* __restrict__ b1,
                                   float* __restrict__ table) {
    int idx = blockIdx.x * blockDim.x + threadIdx.x;   // 0 .. 80*1024-1
    if (idx >= VOCAB * FOURH) return;
    int v = idx >> 10;
    int n = idx & 1023;
    float s = b1[n];
#pragma unroll
    for (int e = 0; e < EMBD; ++e) s += emb[v * EMBD + e] * W1[e * FOURH + n];
    table[v * FOURH + n] = s;
}

__global__ void zero_f32_kernel(float* __restrict__ p, int n) {
    int i = blockIdx.x * blockDim.x + threadIdx.x;
    if (i < n) p[i] = 0.0f;
}
__global__ void zero_f16_kernel(_Float16* __restrict__ p, int n) {
    int i = blockIdx.x * blockDim.x + threadIdx.x;
    if (i < n) p[i] = (_Float16)0.0f;
}

// ---------------------------------------------------------------------------
// Fused LSTM step: gates = [x-part] + h_prev @ WhT^T ; pointwise cell update.
//   grid  = (BATCH/64, HID/64), block = 256 threads (8 waves, wave32).
//   Each wave owns one 16-row m-tile and two 16-col hidden n-tiles, with all
//   four gate groups (i,j,f,o) -> 8 accumulator tiles, no cross-wave traffic.
// Layer 1: xA==nullptr, x-contribution added from `table` via feature lookup.
// Layer 2: table==nullptr, x-contribution is a second K=256 WMMA GEMM vs xA.
// ---------------------------------------------------------------------------
__global__ void __launch_bounds__(256)
lstm_step_kernel(const _Float16* __restrict__ xA,     // [B][256] f16 or null
                 const _Float16* __restrict__ WxT,    // [1024][256] f16 or null
                 const _Float16* __restrict__ hprev,  // [B][256] f16
                 const _Float16* __restrict__ WhT,    // [1024][256] f16
                 const float*    __restrict__ table,  // [V][1024] f32 or null
                 const int*      __restrict__ feat,   // [B][T] or null
                 int t,
                 const float*    __restrict__ bias,   // [1024] f32 or null
                 float*          __restrict__ c,      // [B][256] f32 in/out
                 _Float16*       __restrict__ hnext)  // [B][256] f16 out
{
    const int lane   = threadIdx.x & 31;
    const int wave   = threadIdx.x >> 5;
    const int laneN  = lane & 15;     // N column within 16-wide tile
    const int laneHi = lane >> 4;     // 0/1 : selects K/M halves per ISA layout
    const int kb8    = laneHi * 8;    // A-fragment K base
    const int kb16   = laneHi * 8;    // B-fragment K base

    const int mbase = blockIdx.x * 64;
    const int hbase = blockIdx.y * 64;
    const int mt    = wave & 3;       // m-tile 0..3 (16 rows each)
    const int nt0   = wave >> 2;      // first hidden n-tile (0/1); second is +2

    // 8 accumulator tiles: acc[q*4+g], q in {0,1} selects n-tile nt0+2q, g = gate
    v8f acc[8];
#pragma unroll
    for (int i = 0; i < 8; ++i) acc[i] = zero8();

    // A-fragment row per ISA layout: M = lane & 15
    const int arow = mbase + mt * 16 + laneN;

    // ---- recurrent GEMM: h_prev @ Wh ----
    {
        const _Float16* aBase = hprev + (size_t)arow * HID;
#pragma unroll
        for (int kc = 0; kc < HID; kc += 32) {
            v8h lo = *(const v8h*)(aBase + kc + kb8);
            v8h hi = *(const v8h*)(aBase + kc + kb8 + 16);
            v16h afrag;
#pragma unroll
            for (int e = 0; e < 8; ++e) { afrag[e] = lo[e]; afrag[8 + e] = hi[e]; }
#pragma unroll
            for (int ti = 0; ti < 8; ++ti) {
                const int g   = ti & 3;
                const int nt  = nt0 + (ti >> 2) * 2;
                const int col = g * HID + hbase + nt * 16 + laneN;
                v8h blo = *(const v8h*)(WhT + (size_t)col * HID + kc + kb16);
                v8h bhi = *(const v8h*)(WhT + (size_t)col * HID + kc + kb16 + 16);
                v16h bfrag;
#pragma unroll
                for (int e = 0; e < 8; ++e) { bfrag[e] = blo[e]; bfrag[8 + e] = bhi[e]; }
                acc[ti] = __builtin_amdgcn_wmma_f32_16x16x32_f16(
                    false, afrag, false, bfrag, (short)0, acc[ti], false, false);
            }
        }
    }

    // ---- layer-2 input GEMM: h1_t @ Wx ----
    if (xA != nullptr) {
        const _Float16* aBase = xA + (size_t)arow * HID;
#pragma unroll
        for (int kc = 0; kc < HID; kc += 32) {
            v8h lo = *(const v8h*)(aBase + kc + kb8);
            v8h hi = *(const v8h*)(aBase + kc + kb8 + 16);
            v16h afrag;
#pragma unroll
            for (int e = 0; e < 8; ++e) { afrag[e] = lo[e]; afrag[8 + e] = hi[e]; }
#pragma unroll
            for (int ti = 0; ti < 8; ++ti) {
                const int g   = ti & 3;
                const int nt  = nt0 + (ti >> 2) * 2;
                const int col = g * HID + hbase + nt * 16 + laneN;
                v8h blo = *(const v8h*)(WxT + (size_t)col * HID + kc + kb16);
                v8h bhi = *(const v8h*)(WxT + (size_t)col * HID + kc + kb16 + 16);
                v16h bfrag;
#pragma unroll
                for (int e = 0; e < 8; ++e) { bfrag[e] = blo[e]; bfrag[8 + e] = bhi[e]; }
                acc[ti] = __builtin_amdgcn_wmma_f32_16x16x32_f16(
                    false, afrag, false, bfrag, (short)0, acc[ti], false, false);
            }
        }
    }

    // ---- add x-table (layer 1) or bias (layer 2) ----
    // C/D layout: element r of each tile lives at M = mt*16 + laneHi*8 + r, N = laneN.
    if (table != nullptr) {
#pragma unroll
        for (int r = 0; r < 8; ++r) {
            const int m = mbase + mt * 16 + laneHi * 8 + r;
            const int f = feat[(size_t)m * SEQT + t];
            const float* trow = table + (size_t)f * FOURH;
#pragma unroll
            for (int ti = 0; ti < 8; ++ti) {
                const int g  = ti & 3;
                const int nt = nt0 + (ti >> 2) * 2;
                acc[ti][r] += trow[g * HID + hbase + nt * 16 + laneN];
            }
        }
    } else {
#pragma unroll
        for (int ti = 0; ti < 8; ++ti) {
            const int g  = ti & 3;
            const int nt = nt0 + (ti >> 2) * 2;
            const float bb = bias[g * HID + hbase + nt * 16 + laneN];
#pragma unroll
            for (int r = 0; r < 8; ++r) acc[ti][r] += bb;
        }
    }

    // ---- pointwise LSTM cell (TF gate order i, j, f, o; forget bias = 1.0) ----
#pragma unroll
    for (int q = 0; q < 2; ++q) {
        const int nt   = nt0 + q * 2;
        const v8f gi = acc[q * 4 + 0];
        const v8f gj = acc[q * 4 + 1];
        const v8f gf = acc[q * 4 + 2];
        const v8f go = acc[q * 4 + 3];
#pragma unroll
        for (int r = 0; r < 8; ++r) {
            const int m    = mbase + mt * 16 + laneHi * 8 + r;
            const int hcol = hbase + nt * 16 + laneN;
            const size_t idx = (size_t)m * HID + hcol;
            const float cold = c[idx];
            const float iv = sigf(gi[r]);
            const float jv = tanhf(gj[r]);
            const float fv = sigf(gf[r] + 1.0f);
            const float ov = sigf(go[r]);
            const float cn = cold * fv + iv * jv;
            c[idx]     = cn;
            hnext[idx] = (_Float16)(tanhf(cn) * ov);
        }
    }
}

// ---------------------------------------------------------------------------
// Final dense + log-softmax CE per batch row.  Tiny FLOPs -> plain VALU.
// One block per row: 128 threads; threads 0..79 each own one class.
// ---------------------------------------------------------------------------
__global__ void logits_loss_kernel(const _Float16* __restrict__ h2,
                                   const float* __restrict__ Wd,
                                   const float* __restrict__ bd,
                                   const int*  __restrict__ labels,
                                   float* __restrict__ lossrow) {
    __shared__ float sh[HID];
    __shared__ float slog[VOCAB];
    const int m = blockIdx.x;
    for (int k = threadIdx.x; k < HID; k += blockDim.x)
        sh[k] = (float)h2[(size_t)m * HID + k];
    __syncthreads();
    if (threadIdx.x < VOCAB) {
        const int v = threadIdx.x;
        float s = bd[v];
#pragma unroll 8
        for (int k = 0; k < HID; ++k) s += sh[k] * Wd[k * VOCAB + v];
        slog[v] = s;
    }
    __syncthreads();
    if (threadIdx.x == 0) {
        float mx = slog[0];
        for (int v = 1; v < VOCAB; ++v) mx = fmaxf(mx, slog[v]);
        float se = 0.0f;
        for (int v = 0; v < VOCAB; ++v) se += __expf(slog[v] - mx);
        const int lab = labels[m];
        lossrow[m] = -(slog[lab] - mx - __logf(se));
    }
}

__global__ void reduce_loss_kernel(const float* __restrict__ lossrow,
                                   float* __restrict__ out) {
    __shared__ float s[256];
    float a = 0.0f;
    for (int i = threadIdx.x; i < BATCH; i += 256) a += lossrow[i];
    s[threadIdx.x] = a;
    __syncthreads();
    for (int st = 128; st > 0; st >>= 1) {
        if (threadIdx.x < st) s[threadIdx.x] += s[threadIdx.x + st];
        __syncthreads();
    }
    if (threadIdx.x == 0) out[0] = s[0] / (float)BATCH;
}

// ---------------------------------------------------------------------------
// Host-side orchestration (graph-capture safe: only kernel launches).
// ---------------------------------------------------------------------------
extern "C" void kernel_launch(void* const* d_in, const int* in_sizes, int n_in,
                              void* d_out, int out_size, void* d_ws, size_t ws_size,
                              hipStream_t stream) {
    (void)in_sizes; (void)n_in; (void)out_size; (void)ws_size;

    const int*   features = (const int*)d_in[0];   // [B,T]
    const int*   labels   = (const int*)d_in[1];   // [B]
    const float* emb      = (const float*)d_in[2]; // [V,E]
    const float* W1       = (const float*)d_in[3]; // [E+H, 4H]
    const float* b1       = (const float*)d_in[4]; // [4H]
    const float* W2       = (const float*)d_in[5]; // [2H, 4H]
    const float* b2       = (const float*)d_in[6]; // [4H]
    const float* Wd       = (const float*)d_in[7]; // [H, V]
    const float* bd       = (const float*)d_in[8]; // [V]
    float*       out      = (float*)d_out;

    // --- carve workspace (256B-aligned blocks) ---
    size_t off = 0;
    char* base = (char*)d_ws;
    auto carve = [&](size_t bytes) -> char* {
        char* p = base + off;
        off += (bytes + 255) & ~(size_t)255;
        return p;
    };
    _Float16* W1hT   = (_Float16*)carve((size_t)FOURH * HID * 2);
    _Float16* W2xT   = (_Float16*)carve((size_t)FOURH * HID * 2);
    _Float16* W2hT   = (_Float16*)carve((size_t)FOURH * HID * 2);
    float*    table1 = (float*)   carve((size_t)VOCAB * FOURH * 4);
    float*    c1     = (float*)   carve((size_t)BATCH * HID * 4);
    float*    c2     = (float*)   carve((size_t)BATCH * HID * 4);
    _Float16* h1a    = (_Float16*)carve((size_t)BATCH * HID * 2);
    _Float16* h1b    = (_Float16*)carve((size_t)BATCH * HID * 2);
    _Float16* h2a    = (_Float16*)carve((size_t)BATCH * HID * 2);
    _Float16* h2b    = (_Float16*)carve((size_t)BATCH * HID * 2);
    float*    lrow   = (float*)   carve((size_t)BATCH * 4);

    const int nState = BATCH * HID;   // 1M elements

    // --- prep ---
    zero_f32_kernel<<<(nState + 255) / 256, 256, 0, stream>>>(c1, nState);
    zero_f32_kernel<<<(nState + 255) / 256, 256, 0, stream>>>(c2, nState);
    zero_f16_kernel<<<(nState + 255) / 256, 256, 0, stream>>>(h1a, nState);
    zero_f16_kernel<<<(nState + 255) / 256, 256, 0, stream>>>(h2a, nState);
    prep_weights_kernel<<<(FOURH * HID + 255) / 256, 256, 0, stream>>>(
        W1, W2, W1hT, W2xT, W2hT);
    build_table_kernel<<<(VOCAB * FOURH + 255) / 256, 256, 0, stream>>>(
        emb, W1, b1, table1);

    // --- recurrence: 2 fused WMMA step kernels per timestep ---
    const dim3 grid(BATCH / 64, HID / 64);
    for (int t = 0; t < SEQT; ++t) {
        _Float16* h1p = (t & 1) ? h1b : h1a;
        _Float16* h1n = (t & 1) ? h1a : h1b;
        _Float16* h2p = (t & 1) ? h2b : h2a;
        _Float16* h2n = (t & 1) ? h2a : h2b;

        // layer 1: gates = table1[feat[:,t]] + h1_prev @ W1h
        lstm_step_kernel<<<grid, 256, 0, stream>>>(
            nullptr, nullptr, h1p, W1hT, table1, features, t, nullptr, c1, h1n);
        // layer 2: gates = h1_t @ W2x + h2_prev @ W2h + b2
        lstm_step_kernel<<<grid, 256, 0, stream>>>(
            h1n, W2xT, h2p, W2hT, nullptr, nullptr, t, b2, c2, h2n);
    }

    // t = 79 (odd) writes h2 into h2a
    _Float16* h2last = h2a;
    logits_loss_kernel<<<BATCH, 128, 0, stream>>>(h2last, Wd, bd, labels, lrow);
    reduce_loss_kernel<<<1, 256, 0, stream>>>(lrow, out);
}
